// MEMAE_17910013625042
// MI455X (gfx1250) — matrix operand order repaired and checked by reading, not compiled, not run on hardware
//
#include <hip/hip_runtime.h>
#include <hip/hip_bf16.h>
#include <math.h>

typedef __attribute__((ext_vector_type(16))) _Float16 v16h;
typedef __attribute__((ext_vector_type(8)))  _Float16 v8h;
typedef __attribute__((ext_vector_type(8)))  float    v8f;

#define TPB 256

// ---------------- direct convolution (NCHW, OIHW weights) ----------------
__global__ void conv2d_kernel(const float* __restrict__ in, const float* __restrict__ w,
                              const float* __restrict__ bias, float* __restrict__ out,
                              int N, int Ci, int Hi, int Wi,
                              int Co, int Ho, int Wo, int K, int S, int P) {
  long idx = (long)blockIdx.x * blockDim.x + threadIdx.x;
  long total = (long)N * Co * Ho * Wo;
  if (idx >= total) return;
  int ox = (int)(idx % Wo); long t = idx / Wo;
  int oy = (int)(t % Ho); t /= Ho;
  int co = (int)(t % Co); int n = (int)(t / Co);
  float acc = bias[co];
  const float* wbase = w + (size_t)co * Ci * K * K;
  const float* ibase = in + (size_t)n * Ci * Hi * Wi;
  for (int ci = 0; ci < Ci; ++ci) {
    const float* ich = ibase + (size_t)ci * Hi * Wi;
    const float* wch = wbase + (size_t)ci * K * K;
    for (int ky = 0; ky < K; ++ky) {
      int iy = oy * S - P + ky;
      if (iy < 0 || iy >= Hi) continue;
      for (int kx = 0; kx < K; ++kx) {
        int ix = ox * S - P + kx;
        if (ix < 0 || ix >= Wi) continue;
        acc = fmaf(ich[iy * Wi + ix], wch[ky * K + kx], acc);
      }
    }
  }
  out[idx] = acc;
}

// --------- ConvTranspose2d, gather form (torch weight layout (Ci,Co,K,K)) ---------
__global__ void deconv2d_kernel(const float* __restrict__ in, const float* __restrict__ w,
                                const float* __restrict__ bias, float* __restrict__ out,
                                int N, int Ci, int Hi, int Wi,
                                int Co, int Ho, int Wo, int K, int S, int P) {
  long idx = (long)blockIdx.x * blockDim.x + threadIdx.x;
  long total = (long)N * Co * Ho * Wo;
  if (idx >= total) return;
  int ox = (int)(idx % Wo); long t = idx / Wo;
  int oy = (int)(t % Ho); t /= Ho;
  int co = (int)(t % Co); int n = (int)(t / Co);
  float acc = bias[co];
  const float* ibase = in + (size_t)n * Ci * Hi * Wi;
  for (int ky = 0; ky < K; ++ky) {
    int ty = oy + P - ky;
    if (ty < 0 || (ty % S) != 0) continue;
    int iy = ty / S;
    if (iy >= Hi) continue;
    for (int kx = 0; kx < K; ++kx) {
      int tx = ox + P - kx;
      if (tx < 0 || (tx % S) != 0) continue;
      int ix = tx / S;
      if (ix >= Wi) continue;
      for (int ci = 0; ci < Ci; ++ci) {
        float iv = ibase[((size_t)ci * Hi + iy) * Wi + ix];
        float wv = w[(((size_t)ci * Co + co) * K + ky) * K + kx];
        acc = fmaf(iv, wv, acc);
      }
    }
  }
  out[idx] = acc;
}

// ---------------- BatchNorm (training mode): two-stage deterministic stats ----------------
__global__ void bn_stats_kernel(const float* __restrict__ x, float* __restrict__ partial,
                                int N, int C, int HW, int NS) {
  int c = blockIdx.x;
  int sidx = blockIdx.y;
  long per = (long)N * HW;
  float s = 0.f, s2 = 0.f;
  for (long i = (long)sidx * blockDim.x + threadIdx.x; i < per; i += (long)NS * blockDim.x) {
    long n = i / HW, p = i % HW;
    float v = x[((size_t)n * C + c) * HW + p];
    s += v; s2 += v * v;
  }
  __shared__ float r1[TPB], r2[TPB];
  r1[threadIdx.x] = s; r2[threadIdx.x] = s2;
  __syncthreads();
  for (int o = TPB / 2; o > 0; o >>= 1) {
    if (threadIdx.x < o) { r1[threadIdx.x] += r1[threadIdx.x + o]; r2[threadIdx.x] += r2[threadIdx.x + o]; }
    __syncthreads();
  }
  if (threadIdx.x == 0) {
    partial[((size_t)c * NS + sidx) * 2 + 0] = r1[0];
    partial[((size_t)c * NS + sidx) * 2 + 1] = r2[0];
  }
}

__global__ void bn_finalize_kernel(const float* __restrict__ partial,
                                   const float* __restrict__ g, const float* __restrict__ b,
                                   float* __restrict__ scale, float* __restrict__ shift,
                                   int NS, long count) {
  int c = blockIdx.x;
  __shared__ float r1[64], r2[64];
  float s = 0.f, s2 = 0.f;
  for (int i = threadIdx.x; i < NS; i += blockDim.x) {
    s  += partial[((size_t)c * NS + i) * 2 + 0];
    s2 += partial[((size_t)c * NS + i) * 2 + 1];
  }
  r1[threadIdx.x] = s; r2[threadIdx.x] = s2;
  __syncthreads();
  for (int o = 32; o > 0; o >>= 1) {
    if (threadIdx.x < o) { r1[threadIdx.x] += r1[threadIdx.x + o]; r2[threadIdx.x] += r2[threadIdx.x + o]; }
    __syncthreads();
  }
  if (threadIdx.x == 0) {
    float inv = 1.f / (float)count;
    float m = r1[0] * inv;
    float var = r2[0] * inv - m * m;
    float sc = g[c] * rsqrtf(var + 1e-5f);
    scale[c] = sc;
    shift[c] = b[c] - m * sc;
  }
}

__global__ void bn_apply_relu_kernel(float* __restrict__ x, const float* __restrict__ scale,
                                     const float* __restrict__ shift, long total, int C, int HW) {
  long idx = (long)blockIdx.x * blockDim.x + threadIdx.x;
  if (idx >= total) return;
  int c = (int)((idx / HW) % C);
  x[idx] = fmaxf(fmaf(x[idx], scale[c], shift[c]), 0.f);
}

// ---------------- row L2 norms ----------------
__global__ void rownorm_kernel(const float* __restrict__ x, float* __restrict__ out, int F) {
  int r = blockIdx.x;
  __shared__ float red[TPB];
  const float* row = x + (size_t)r * F;
  float s = 0.f;
  for (int i = threadIdx.x; i < F; i += TPB) { float v = row[i]; s += v * v; }
  red[threadIdx.x] = s; __syncthreads();
  for (int o = TPB / 2; o > 0; o >>= 1) {
    if (threadIdx.x < o) red[threadIdx.x] += red[threadIdx.x + o];
    __syncthreads();
  }
  if (threadIdx.x == 0) out[r] = sqrtf(red[0]);
}

// ---------------- f32 -> f16 convert ----------------
__global__ void cvt_f16_kernel(const float* __restrict__ x, _Float16* __restrict__ y, long n) {
  long i = (long)blockIdx.x * blockDim.x + threadIdx.x;
  if (i < n) y[i] = (_Float16)x[i];
}

// transpose memory (M x F, f32) -> memT (F x Kp, f16) with zero pad for cols >= M
__global__ void transpose_pad_kernel(const float* __restrict__ mem, _Float16* __restrict__ mt,
                                     int M, int F, int Kp) {
  long idx = (long)blockIdx.x * blockDim.x + threadIdx.x;
  long total = (long)F * Kp;
  if (idx >= total) return;
  int m = (int)(idx % Kp);
  int f = (int)(idx / Kp);
  mt[idx] = (m < M) ? (_Float16)mem[(size_t)m * F + f] : (_Float16)0.f;
}

// ---------------- WMMA NT GEMM: C[i,j] = sum_k A[i,k] * B[j,k] ----------------
// A: Mrows x K (lda), B: Ncols x K (ldb), both f16 row-major; C f32 (ldc).
// One wave per 16x16 tile; K multiple of 32; M,N multiples of 16.
__global__ void __launch_bounds__(256)
wmma_gemm_nt_kernel(const _Float16* __restrict__ A, int lda,
                    const _Float16* __restrict__ B, int ldb,
                    float* __restrict__ C, int ldc,
                    int K, int tilesN, int totalTiles) {
  int wave = blockIdx.x * (blockDim.x >> 5) + (threadIdx.x >> 5);
  if (wave >= totalTiles) return;            // wave-uniform: EXEC stays all-ones for WMMA
  int tm = wave / tilesN;
  int tn = wave % tilesN;
  int lane = threadIdx.x & 31;
  int hi = lane >> 4;                         // half-wave select
  int lr = lane & 15;

  // A lane layout (16x32 f16): elems 0..15 -> K = k0+16*hi .. +15
  const _Float16* arow = A + (size_t)(tm * 16 + lr) * lda + (hi << 4);
  // B lane layout (32x16 f16): elems 0..15 -> K = k0+16*hi .. +15, column N = lr
  const _Float16* brow = B + (size_t)(tn * 16 + lr) * ldb + (hi << 4);

  v8f acc = {};
  for (int k0 = 0; k0 < K; k0 += 32) {
    v8h a0 = *(const v8h*)(arow + k0);
    v8h a1 = *(const v8h*)(arow + k0 + 8);
    union { v16h v; v8h h[2]; } au;
    au.h[0] = a0; au.h[1] = a1;
    v16h bv = *(const v16h*)(brow + k0);
    acc = __builtin_amdgcn_wmma_f32_16x16x32_f16(
        /*neg_a=*/false, au.v, /*neg_b=*/false, bv,
        /*c_mod=*/(short)0, acc, /*reuse_a=*/false, /*reuse_b=*/false);
  }
  // C/D layout: lane<16 -> N=lane, rows r=0..7 ; lane>=16 -> rows 8..15
  float* crow = C + (size_t)(tm * 16 + (hi << 3)) * ldc + tn * 16 + lr;
#pragma unroll
  for (int r = 0; r < 8; ++r) crow[(size_t)r * ldc] = acc[r];
}

// ---------------- fused cosine-softmax + hard shrink + L1 renorm ----------------
#define MMEM 2000
__global__ void __launch_bounds__(256)
softmax_shrink_kernel(const float* __restrict__ sim, const float* __restrict__ zn,
                      const float* __restrict__ mn, _Float16* __restrict__ w16,
                      int Kp, float tshrink) {
  int b = blockIdx.x;
  __shared__ float sv[MMEM];
  __shared__ float red[TPB];
  const float znb = zn[b];
  const float* srow = sim + (size_t)b * MMEM;
  int tid = threadIdx.x;

  float lmax = -3.4e38f;
  for (int m = tid; m < MMEM; m += TPB) {
    float d = fmaxf(znb * mn[m], 1e-8f);
    float s = srow[m] / d;
    sv[m] = s;
    lmax = fmaxf(lmax, s);
  }
  red[tid] = lmax; __syncthreads();
  for (int o = TPB / 2; o > 0; o >>= 1) {
    if (tid < o) red[tid] = fmaxf(red[tid], red[tid + o]);
    __syncthreads();
  }
  float gmax = red[0]; __syncthreads();

  float lsum = 0.f;
  for (int m = tid; m < MMEM; m += TPB) {
    float e = expf(sv[m] - gmax);
    sv[m] = e;
    lsum += e;
  }
  red[tid] = lsum; __syncthreads();
  for (int o = TPB / 2; o > 0; o >>= 1) {
    if (tid < o) red[tid] += red[tid + o];
    __syncthreads();
  }
  float ginvsum = 1.f / red[0]; __syncthreads();

  float lsa = 0.f;
  for (int m = tid; m < MMEM; m += TPB) {
    float w = sv[m] * ginvsum;
    float u = w - tshrink;
    float w2 = fmaxf(u, 0.f) * w / (fabsf(u) + 0.01f);
    sv[m] = w2;
    lsa += w2;     // w2 >= 0, so L1 sum == plain sum
  }
  red[tid] = lsa; __syncthreads();
  for (int o = TPB / 2; o > 0; o >>= 1) {
    if (tid < o) red[tid] += red[tid + o];
    __syncthreads();
  }
  float ginvl1 = 1.f / red[0]; __syncthreads();

  _Float16* wrow = w16 + (size_t)b * Kp;
  for (int m = tid; m < Kp; m += TPB)
    wrow[m] = (m < MMEM) ? (_Float16)(sv[m] * ginvl1) : (_Float16)0.f;
}

// ---------------- sigmoid (in place) ----------------
__global__ void sigmoid_kernel(float* __restrict__ x, long n) {
  long i = (long)blockIdx.x * blockDim.x + threadIdx.x;
  if (i < n) x[i] = 1.f / (1.f + expf(-x[i]));
}

static inline unsigned nblk(long total) { return (unsigned)((total + TPB - 1) / TPB); }

extern "C" void kernel_launch(void* const* d_in, const int* in_sizes, int n_in,
                              void* d_out, int out_size, void* d_ws, size_t ws_size,
                              hipStream_t stream) {
  (void)in_sizes; (void)n_in; (void)ws_size;
  const int NB = 512, M = 2000, F = 2304, Kp = 2016, NS = 64;

  const float* x     = (const float*)d_in[0];
  const float* c1_w  = (const float*)d_in[1];  const float* c1_b  = (const float*)d_in[2];
  const float* bn1_g = (const float*)d_in[3];  const float* bn1_b = (const float*)d_in[4];
  const float* c2_w  = (const float*)d_in[5];  const float* c2_b  = (const float*)d_in[6];
  const float* bn2_g = (const float*)d_in[7];  const float* bn2_b = (const float*)d_in[8];
  const float* c3_w  = (const float*)d_in[9];  const float* c3_b  = (const float*)d_in[10];
  const float* bn3_g = (const float*)d_in[11]; const float* bn3_b = (const float*)d_in[12];
  const float* c4_w  = (const float*)d_in[13]; const float* c4_b  = (const float*)d_in[14];
  const float* bn4_g = (const float*)d_in[15]; const float* bn4_b = (const float*)d_in[16];
  const float* memry = (const float*)d_in[17];
  const float* d0_w  = (const float*)d_in[18]; const float* d0_b  = (const float*)d_in[19];
  const float* dbn0_g= (const float*)d_in[20]; const float* dbn0_b= (const float*)d_in[21];
  const float* d1_w  = (const float*)d_in[22]; const float* d1_b  = (const float*)d_in[23];
  const float* dbn1_g= (const float*)d_in[24]; const float* dbn1_b= (const float*)d_in[25];
  const float* d2_w  = (const float*)d_in[26]; const float* d2_b  = (const float*)d_in[27];
  const float* dbn2_g= (const float*)d_in[28]; const float* dbn2_b= (const float*)d_in[29];
  const float* d3_w  = (const float*)d_in[30]; const float* d3_b  = (const float*)d_in[31];

  // ---- workspace carve-out (256B aligned) ----
  size_t off = 0;
  char* wsb = (char*)d_ws;
  auto alloc = [&](size_t bytes) -> void* {
    void* p = wsb + off;
    off += (bytes + 255) & ~(size_t)255;
    return p;
  };
  float*    h1    = (float*)alloc((size_t)NB * 16 * 49 * 49 * 4);  // also decoder g2
  float*    h2    = (float*)alloc((size_t)NB * 32 * 25 * 25 * 4);  // also decoder g1
  float*    h3    = (float*)alloc((size_t)NB * 64 * 13 * 13 * 4);  // also decoder g0
  float*    z     = (float*)alloc((size_t)NB * F * 4);
  float*    zhat  = (float*)alloc((size_t)NB * F * 4);
  _Float16* z16   = (_Float16*)alloc((size_t)NB * F * 2);
  _Float16* mem16 = (_Float16*)alloc((size_t)M * F * 2);
  _Float16* memT16= (_Float16*)alloc((size_t)F * Kp * 2);
  float*    sim   = (float*)alloc((size_t)NB * M * 4);
  _Float16* w16   = (_Float16*)alloc((size_t)NB * Kp * 2);
  float*    zn    = (float*)alloc((size_t)NB * 4);
  float*    mn    = (float*)alloc((size_t)M * 4);
  float*    bnpart= (float*)alloc((size_t)64 * NS * 2 * 4);
  float*    bnsc  = (float*)alloc(64 * 4);
  float*    bnsh  = (float*)alloc(64 * 4);

  auto bn = [&](float* buf, const float* g, const float* bb, int C, int HW) {
    long count = (long)NB * HW;
    dim3 gs((unsigned)C, (unsigned)NS);
    bn_stats_kernel<<<gs, TPB, 0, stream>>>(buf, bnpart, NB, C, HW, NS);
    bn_finalize_kernel<<<C, 64, 0, stream>>>(bnpart, g, bb, bnsc, bnsh, NS, count);
    long total = (long)NB * C * HW;
    bn_apply_relu_kernel<<<nblk(total), TPB, 0, stream>>>(buf, bnsc, bnsh, total, C, HW);
  };

  // ---------------- Encoder ----------------
  conv2d_kernel<<<nblk((long)NB*16*49*49), TPB, 0, stream>>>(x,  c1_w, c1_b, h1, NB, 1,  96, 96, 16, 49, 49, 1, 2, 1);
  bn(h1, bn1_g, bn1_b, 16, 49*49);
  conv2d_kernel<<<nblk((long)NB*32*25*25), TPB, 0, stream>>>(h1, c2_w, c2_b, h2, NB, 16, 49, 49, 32, 25, 25, 3, 2, 1);
  bn(h2, bn2_g, bn2_b, 32, 25*25);
  conv2d_kernel<<<nblk((long)NB*64*13*13), TPB, 0, stream>>>(h2, c3_w, c3_b, h3, NB, 32, 25, 25, 64, 13, 13, 3, 2, 1);
  bn(h3, bn3_g, bn3_b, 64, 13*13);
  conv2d_kernel<<<nblk((long)NB*64*6*6),   TPB, 0, stream>>>(h3, c4_w, c4_b, z,  NB, 64, 13, 13, 64, 6,  6,  3, 2, 0);
  bn(z, bn4_g, bn4_b, 64, 6*6);

  // ---------------- Memory addressing ----------------
  rownorm_kernel<<<NB, TPB, 0, stream>>>(z, zn, F);
  rownorm_kernel<<<M,  TPB, 0, stream>>>(memry, mn, F);
  cvt_f16_kernel<<<nblk((long)NB*F), TPB, 0, stream>>>(z, z16, (long)NB*F);
  cvt_f16_kernel<<<nblk((long)M*F),  TPB, 0, stream>>>(memry, mem16, (long)M*F);
  transpose_pad_kernel<<<nblk((long)F*Kp), TPB, 0, stream>>>(memry, memT16, M, F, Kp);

  // sim = z @ mem^T : (512 x 2000), K = 2304
  {
    int tilesN = M / 16, tiles = (NB / 16) * tilesN;
    wmma_gemm_nt_kernel<<<(tiles + 7) / 8, 256, 0, stream>>>(z16, F, mem16, F, sim, M, F, tilesN, tiles);
  }

  // softmax over M + hard shrink + L1 renorm -> f16 weights (padded K to 2016)
  softmax_shrink_kernel<<<NB, TPB, 0, stream>>>(sim, zn, mn, w16, Kp, 1.f / (float)M);

  // zhat = w @ mem : (512 x 2304), K = 2016 (zero-padded 2000)
  {
    int tilesN = F / 16, tiles = (NB / 16) * tilesN;
    wmma_gemm_nt_kernel<<<(tiles + 7) / 8, 256, 0, stream>>>(w16, Kp, memT16, Kp, zhat, F, Kp, tilesN, tiles);
  }

  // ---------------- Decoder ----------------
  float* g0 = h3; float* g1 = h2; float* g2 = h1;
  deconv2d_kernel<<<nblk((long)NB*64*13*13), TPB, 0, stream>>>(zhat, d0_w, d0_b, g0, NB, 64, 6,  6,  64, 13, 13, 3, 2, 0);
  bn(g0, dbn0_g, dbn0_b, 64, 13*13);
  deconv2d_kernel<<<nblk((long)NB*32*25*25), TPB, 0, stream>>>(g0, d1_w, d1_b, g1, NB, 64, 13, 13, 32, 25, 25, 3, 2, 1);
  bn(g1, dbn1_g, dbn1_b, 32, 25*25);
  deconv2d_kernel<<<nblk((long)NB*16*49*49), TPB, 0, stream>>>(g1, d2_w, d2_b, g2, NB, 32, 25, 25, 16, 49, 49, 2, 2, 1);
  bn(g2, dbn2_g, dbn2_b, 16, 49*49);

  float* outp = (float*)d_out;
  deconv2d_kernel<<<nblk((long)NB*1*98*98), TPB, 0, stream>>>(g2, d3_w, d3_b, outp, NB, 16, 49, 49, 1, 98, 98, 2, 2, 0);
  sigmoid_kernel<<<nblk((long)out_size), TPB, 0, stream>>>(outp, (long)out_size);
}
